// SwinBlock_73847667687717
// MI455X (gfx1250) — compile-verified, matrix operand order checked by reading, not run on hardware
//
#include <hip/hip_runtime.h>
#include <hip/hip_bf16.h>

typedef __attribute__((ext_vector_type(16))) __bf16 v16bf;
typedef __attribute__((ext_vector_type(8)))  __bf16 v8bf;
typedef __attribute__((ext_vector_type(8)))  float  v8f;

#define TOKENS 100352L   // 32*56*56
#define CDIM   512
#define NHEAD  16
#define DHEAD  32
#define NWIN   2048L     // 32 * 8 * 8 windows
#define NTOK   49
#define NPAD   64

// ---------------------------------------------------------------------------
// Fragment loader for 16x32 bf16 A-tiles (and B-tiles from a transposed BT
// matrix, which has the identical per-lane pattern):
//   lane l (0..15)  holds row (l&15), K = kb..kb+7 and kb+16..kb+23, kb = 0
//   lane l (16..31) holds row (l&15), same with kb = 8
// Two 16B vector loads per lane (contiguous within a row).
// ---------------------------------------------------------------------------
__device__ __forceinline__ v16bf load_frag_b16(const __bf16* base, int ld) {
  int l = threadIdx.x & 31;
  const __bf16* p = base + (long)(l & 15) * ld + ((l >> 4) << 3);
  v8bf lo = *(const v8bf*)(p);
  v8bf hi = *(const v8bf*)(p + 16);
  v16bf r;
#pragma unroll
  for (int i = 0; i < 8; ++i) { r[i] = lo[i]; r[i + 8] = hi[i]; }
  return r;
}

__device__ __forceinline__ v8f zero_v8f() {
  v8f z;
#pragma unroll
  for (int i = 0; i < 8; ++i) z[i] = 0.f;
  return z;
}

#define WMMA_BF16(A, B, Cacc) \
  __builtin_amdgcn_wmma_f32_16x16x32_bf16(false, (A), false, (B), (short)0, (Cacc), false, false)

// ---------------------------------------------------------------------------
// Wave-level GEMM core: one wave computes a 32x64 f32 tile of A(MxK) x BT^T.
// A is row-major bf16 (lda), BT is the transposed B (N x K, row-major, ldb).
// C/D layout: element r of acc[mi][ni] for lane l is
//   M = mBase + mi*16 + r + ((l>>4)<<3),  N = nBase + ni*16 + (l&15)
// ---------------------------------------------------------------------------
__device__ __forceinline__ void wave_gemm_32x64(const __bf16* __restrict__ A, int lda,
                                                const __bf16* __restrict__ BT, int ldb,
                                                long mBase, long nBase, int K,
                                                v8f acc[2][4]) {
#pragma unroll
  for (int mi = 0; mi < 2; ++mi)
#pragma unroll
    for (int ni = 0; ni < 4; ++ni) acc[mi][ni] = zero_v8f();

  for (int k0 = 0; k0 < K; k0 += 32) {
    v16bf a0 = load_frag_b16(A + (mBase     ) * lda + k0, lda);
    v16bf a1 = load_frag_b16(A + (mBase + 16) * lda + k0, lda);
    v16bf b0 = load_frag_b16(BT + (nBase     ) * ldb + k0, ldb);
    v16bf b1 = load_frag_b16(BT + (nBase + 16) * ldb + k0, ldb);
    v16bf b2 = load_frag_b16(BT + (nBase + 32) * ldb + k0, ldb);
    v16bf b3 = load_frag_b16(BT + (nBase + 48) * ldb + k0, ldb);
    acc[0][0] = WMMA_BF16(a0, b0, acc[0][0]);
    acc[0][1] = WMMA_BF16(a0, b1, acc[0][1]);
    acc[0][2] = WMMA_BF16(a0, b2, acc[0][2]);
    acc[0][3] = WMMA_BF16(a0, b3, acc[0][3]);
    acc[1][0] = WMMA_BF16(a1, b0, acc[1][0]);
    acc[1][1] = WMMA_BF16(a1, b1, acc[1][1]);
    acc[1][2] = WMMA_BF16(a1, b2, acc[1][2]);
    acc[1][3] = WMMA_BF16(a1, b3, acc[1][3]);
  }
}

// ---------------------------------------------------------------------------
// Utility kernels
// ---------------------------------------------------------------------------
__global__ void fill_zero_u4_kernel(uint4* __restrict__ p, long n4) {
  long i = (long)blockIdx.x * blockDim.x + threadIdx.x;
  if (i < n4) p[i] = make_uint4(0u, 0u, 0u, 0u);
}

// WT[n*K + k] = bf16(W[k*N + n])
__global__ void convert_transpose_kernel(const float* __restrict__ W,
                                         __bf16* __restrict__ WT, int K, int N) {
  long i = (long)blockIdx.x * blockDim.x + threadIdx.x;
  if (i >= (long)K * N) return;
  int k = (int)(i / N);
  int n = (int)(i - (long)k * N);
  WT[(long)n * K + k] = (__bf16)W[i];
}

// ---------------------------------------------------------------------------
// LayerNorm (+ optional window partition) -> bf16.  One wave per token.
// partitioned=1: output row t is in window-partitioned order, input row mapped.
// ---------------------------------------------------------------------------
__global__ void ln_to_bf16_kernel(const float* __restrict__ xin,
                                  const float* __restrict__ gw,
                                  const float* __restrict__ bw,
                                  __bf16* __restrict__ outp,
                                  int partitioned) {
  int lane = threadIdx.x & 31;
  long t = (long)blockIdx.x * (blockDim.x >> 5) + (threadIdx.x >> 5);
  long src;
  if (partitioned) {
    long w = t / 49;
    int n = (int)(t - w * 49);
    int bb = (int)(w >> 6);
    int wy = ((int)w >> 3) & 7;
    int wx = (int)w & 7;
    int iy = n / 7, ix = n - iy * 7;
    src = ((long)bb * 56 + (wy * 7 + iy)) * 56 + (wx * 7 + ix);
  } else {
    src = t;
  }
  const float* row = xin + src * CDIM;
  float vals[16];
  float s = 0.f, ss = 0.f;
#pragma unroll
  for (int j = 0; j < 16; ++j) {
    float v = row[lane + 32 * j];
    vals[j] = v;
    s += v;
    ss += v * v;
  }
#pragma unroll
  for (int off = 16; off >= 1; off >>= 1) {
    s  += __shfl_xor(s, off, 32);
    ss += __shfl_xor(ss, off, 32);
  }
  float mu = s * (1.f / 512.f);
  float var = ss * (1.f / 512.f) - mu * mu;
  float rstd = rsqrtf(var + 1e-5f);
  __bf16* orow = outp + t * CDIM;
#pragma unroll
  for (int j = 0; j < 16; ++j) {
    int c = lane + 32 * j;
    orow[c] = (__bf16)((vals[j] - mu) * rstd * gw[c] + bw[c]);
  }
}

// ---------------------------------------------------------------------------
// QKV GEMM: xw[100352,512] @ qkv_wT -> scatter into padded per-(win,head)
// staging: q_s/k_s [win][head][64][32] bf16 (q pre-scaled), vT_s [win][head][32][64].
// ---------------------------------------------------------------------------
__global__ void gemm_qkv_kernel(const __bf16* __restrict__ xw,
                                const __bf16* __restrict__ wT,
                                const float* __restrict__ bias,
                                __bf16* __restrict__ q_s,
                                __bf16* __restrict__ k_s,
                                __bf16* __restrict__ vT_s) {
  int wave = threadIdx.x >> 5, lane = threadIdx.x & 31;
  long mBase = ((long)blockIdx.y * 8 + wave) * 32;
  long nBase = (long)blockIdx.x * 64;
  v8f acc[2][4];
  wave_gemm_32x64(xw, CDIM, wT, CDIM, mBase, nBase, CDIM, acc);
  const float scale = 0.17677669529663687f;  // 1/sqrt(32)
#pragma unroll
  for (int mi = 0; mi < 2; ++mi)
#pragma unroll
    for (int ni = 0; ni < 4; ++ni)
#pragma unroll
      for (int r = 0; r < 8; ++r) {
        long t = mBase + mi * 16 + r + ((lane >> 4) << 3);
        int col = (int)nBase + ni * 16 + (lane & 15);
        float v = acc[mi][ni][r] + bias[col];
        int which = col >> 9;
        int c = col & 511;
        int h = c >> 5;
        int d = c & 31;
        long w = t / 49;
        int n = (int)(t - w * 49);
        long wh = w * NHEAD + h;
        if (which == 0)      q_s[(wh * NPAD + n) * DHEAD + d] = (__bf16)(v * scale);
        else if (which == 1) k_s[(wh * NPAD + n) * DHEAD + d] = (__bf16)v;
        else                 vT_s[(wh * DHEAD + d) * NPAD + n] = (__bf16)v;
      }
}

// ---------------------------------------------------------------------------
// Attention: one block (4 waves) per (window, head).
// S = Q K^T (WMMA), + relative-position bias, softmax, O = P V (WMMA via LDS).
// ---------------------------------------------------------------------------
__global__ void attn_kernel(const __bf16* __restrict__ q_s,
                            const __bf16* __restrict__ k_s,
                            const __bf16* __restrict__ vT_s,
                            const float* __restrict__ bias_table,
                            __bf16* __restrict__ attn_out) {
  long wh = blockIdx.x;
  long win = wh >> 4;
  int h = (int)(wh & 15);
  const __bf16* Q = q_s + wh * (NPAD * DHEAD);
  const __bf16* K = k_s + wh * (NPAD * DHEAD);
  const __bf16* V = vT_s + wh * (DHEAD * NPAD);

  int wave = threadIdx.x >> 5;
  int lane = threadIdx.x & 31;
  int mBase = wave * 16;

  __shared__ __bf16 Pld[64][72];  // padded to dodge LDS bank conflicts

  // ---- S = Q K^T : 16x64 strip per wave, K=32 (= dh) in one WMMA step ----
  v16bf qa = load_frag_b16(Q + (long)mBase * DHEAD, DHEAD);
  v8f s[4];
#pragma unroll
  for (int ni = 0; ni < 4; ++ni) {
    v16bf kb = load_frag_b16(K + (long)(ni * 16) * DHEAD, DHEAD);
    s[ni] = WMMA_BF16(qa, kb, zero_v8f());
  }

  // ---- relative position bias + padding mask ----
#pragma unroll
  for (int ni = 0; ni < 4; ++ni)
#pragma unroll
    for (int r = 0; r < 8; ++r) {
      int qi = mBase + r + ((lane >> 4) << 3);
      int ki = ni * 16 + (lane & 15);
      float b;
      if (ki >= NTOK) {
        b = -1e30f;                    // mask padded keys
      } else if (qi >= NTOK) {
        b = 0.f;                       // padded query rows: never read back
      } else {
        int qy = qi / 7, qx = qi - qy * 7;
        int ky = ki / 7, kx = ki - ky * 7;
        int rel = (qy - ky + 6) * 13 + (qx - kx + 6);
        b = bias_table[rel * NHEAD + h];
      }
      s[ni][r] += b;
    }

  // ---- softmax over 64 columns (per row: 4 regs x 16 lanes of a half-wave) ----
#pragma unroll
  for (int r = 0; r < 8; ++r) {
    float m = fmaxf(fmaxf(s[0][r], s[1][r]), fmaxf(s[2][r], s[3][r]));
#pragma unroll
    for (int off = 1; off < 16; off <<= 1) m = fmaxf(m, __shfl_xor(m, off, 32));
    float sum = 0.f;
#pragma unroll
    for (int ni = 0; ni < 4; ++ni) {
      float e = __expf(s[ni][r] - m);
      s[ni][r] = e;
      sum += e;
    }
#pragma unroll
    for (int off = 1; off < 16; off <<= 1) sum += __shfl_xor(sum, off, 32);
    float inv = 1.f / sum;
    int row = mBase + r + ((lane >> 4) << 3);
#pragma unroll
    for (int ni = 0; ni < 4; ++ni) {
      int col = ni * 16 + (lane & 15);
      Pld[row][col] = (__bf16)(s[ni][r] * inv);
    }
  }
  __syncthreads();

  // ---- O = P V : 16x32 strip per wave, K=64 in two steps ----
  v8f o[2] = {zero_v8f(), zero_v8f()};
#pragma unroll
  for (int ks = 0; ks < 2; ++ks) {
    v16bf pa = load_frag_b16(&Pld[mBase][ks * 32], 72);
#pragma unroll
    for (int ni = 0; ni < 2; ++ni) {
      v16bf vb = load_frag_b16(V + (long)(ni * 16) * NPAD + ks * 32, NPAD);
      o[ni] = WMMA_BF16(pa, vb, o[ni]);
    }
  }

  // ---- write attn_out[t, h*32+d] (token-major bf16), drop padded rows ----
#pragma unroll
  for (int ni = 0; ni < 2; ++ni)
#pragma unroll
    for (int r = 0; r < 8; ++r) {
      int n = mBase + r + ((lane >> 4) << 3);
      int d = ni * 16 + (lane & 15);
      if (n < NTOK)
        attn_out[(win * NTOK + n) * CDIM + h * DHEAD + d] = (__bf16)o[ni][r];
    }
}

// ---------------------------------------------------------------------------
// Proj GEMM + bias + residual, with window-unpartition scatter to f32 out.
// ---------------------------------------------------------------------------
__global__ void gemm_proj_kernel(const __bf16* __restrict__ ain,
                                 const __bf16* __restrict__ wT,
                                 const float* __restrict__ bias,
                                 const float* __restrict__ x_in,
                                 float* __restrict__ xout) {
  int wave = threadIdx.x >> 5, lane = threadIdx.x & 31;
  long mBase = ((long)blockIdx.y * 8 + wave) * 32;
  long nBase = (long)blockIdx.x * 64;
  v8f acc[2][4];
  wave_gemm_32x64(ain, CDIM, wT, CDIM, mBase, nBase, CDIM, acc);
#pragma unroll
  for (int mi = 0; mi < 2; ++mi)
#pragma unroll
    for (int ni = 0; ni < 4; ++ni)
#pragma unroll
      for (int r = 0; r < 8; ++r) {
        long t = mBase + mi * 16 + r + ((lane >> 4) << 3);
        int col = (int)nBase + ni * 16 + (lane & 15);
        long w = t / 49;
        int n = (int)(t - w * 49);
        int bb = (int)(w >> 6);
        int wy = ((int)w >> 3) & 7;
        int wx = (int)w & 7;
        int iy = n / 7, ix = n - iy * 7;
        long idx = (((long)bb * 56 + (wy * 7 + iy)) * 56 + (wx * 7 + ix)) * CDIM + col;
        xout[idx] = x_in[idx] + acc[mi][ni][r] + bias[col];
      }
}

// ---------------------------------------------------------------------------
// MLP1 GEMM + bias + exact GELU -> bf16 hidden [tokens, 2048]
// ---------------------------------------------------------------------------
__global__ void gemm_mlp1_kernel(const __bf16* __restrict__ hin,
                                 const __bf16* __restrict__ wT,
                                 const float* __restrict__ bias,
                                 __bf16* __restrict__ gout) {
  int wave = threadIdx.x >> 5, lane = threadIdx.x & 31;
  long mBase = ((long)blockIdx.y * 8 + wave) * 32;
  long nBase = (long)blockIdx.x * 64;
  v8f acc[2][4];
  wave_gemm_32x64(hin, CDIM, wT, CDIM, mBase, nBase, CDIM, acc);
#pragma unroll
  for (int mi = 0; mi < 2; ++mi)
#pragma unroll
    for (int ni = 0; ni < 4; ++ni)
#pragma unroll
      for (int r = 0; r < 8; ++r) {
        long t = mBase + mi * 16 + r + ((lane >> 4) << 3);
        int col = (int)nBase + ni * 16 + (lane & 15);
        float v = acc[mi][ni][r] + bias[col];
        float ge = 0.5f * v * (1.f + erff(v * 0.70710678118654752f));
        gout[t * 2048 + col] = (__bf16)ge;
      }
}

// ---------------------------------------------------------------------------
// MLP2 GEMM + bias + residual accumulate into f32 out (natural token order)
// ---------------------------------------------------------------------------
__global__ void gemm_mlp2_kernel(const __bf16* __restrict__ gin,
                                 const __bf16* __restrict__ wT,
                                 const float* __restrict__ bias,
                                 float* __restrict__ outp) {
  int wave = threadIdx.x >> 5, lane = threadIdx.x & 31;
  long mBase = ((long)blockIdx.y * 8 + wave) * 32;
  long nBase = (long)blockIdx.x * 64;
  v8f acc[2][4];
  wave_gemm_32x64(gin, 2048, wT, 2048, mBase, nBase, 2048, acc);
#pragma unroll
  for (int mi = 0; mi < 2; ++mi)
#pragma unroll
    for (int ni = 0; ni < 4; ++ni)
#pragma unroll
      for (int r = 0; r < 8; ++r) {
        long t = mBase + mi * 16 + r + ((lane >> 4) << 3);
        int col = (int)nBase + ni * 16 + (lane & 15);
        long idx = t * CDIM + col;
        outp[idx] = outp[idx] + acc[mi][ni][r] + bias[col];
      }
}

// ---------------------------------------------------------------------------
// Host launcher
// ---------------------------------------------------------------------------
extern "C" void kernel_launch(void* const* d_in, const int* in_sizes, int n_in,
                              void* d_out, int out_size, void* d_ws, size_t ws_size,
                              hipStream_t stream) {
  const float* x         = (const float*)d_in[0];
  const float* ln1_g     = (const float*)d_in[1];
  const float* ln1_b     = (const float*)d_in[2];
  const float* qkv_w     = (const float*)d_in[3];
  const float* qkv_b     = (const float*)d_in[4];
  const float* proj_w    = (const float*)d_in[5];
  const float* proj_b    = (const float*)d_in[6];
  const float* bias_tab  = (const float*)d_in[7];
  const float* ln2_g     = (const float*)d_in[8];
  const float* ln2_b     = (const float*)d_in[9];
  const float* mlp_w1    = (const float*)d_in[10];
  const float* mlp_b1    = (const float*)d_in[11];
  const float* mlp_w2    = (const float*)d_in[12];
  const float* mlp_b2    = (const float*)d_in[13];
  float* out = (float*)d_out;

  char* ws = (char*)d_ws;
  // R0: tokens x 512 bf16 (xw -> attn_out -> ln2 hidden), 102,760,448 B
  __bf16* r0 = (__bf16*)ws;
  // R1: q/k/vT staging (402.7 MB) reused later as MLP hidden (411.0 MB)
  __bf16* r1 = (__bf16*)(ws + 102760448L);
  __bf16* q_s  = r1;
  __bf16* k_s  = r1 + 67108864L;   // 2048*16*64*32
  __bf16* vT_s = r1 + 134217728L;
  __bf16* gbuf = r1;               // tokens x 2048 bf16
  // Weights (transposed bf16), 6.3 MB
  __bf16* wq = (__bf16*)(ws + 102760448L + 411041792L);
  __bf16* wp = wq + 786432L;
  __bf16* w1 = wp + 262144L;
  __bf16* w2 = w1 + 1048576L;

  // 0) zero q/k/vT staging (padded rows must be exactly 0 each call)
  {
    long n4 = 402653184L / 16L;  // uint4 count
    fill_zero_u4_kernel<<<(unsigned)(n4 / 256), 256, 0, stream>>>((uint4*)r1, n4);
  }
  // 1) weight conversion (f32 -> transposed bf16)
  convert_transpose_kernel<<<3072, 256, 0, stream>>>(qkv_w,  wq, 512, 1536);
  convert_transpose_kernel<<<1024, 256, 0, stream>>>(proj_w, wp, 512, 512);
  convert_transpose_kernel<<<4096, 256, 0, stream>>>(mlp_w1, w1, 512, 2048);
  convert_transpose_kernel<<<4096, 256, 0, stream>>>(mlp_w2, w2, 2048, 512);

  // 2) LN1 + window partition -> bf16 (wave per token, 8 per block)
  ln_to_bf16_kernel<<<12544, 256, 0, stream>>>(x, ln1_g, ln1_b, r0, 1);

  // 3) QKV GEMM: M=100352, K=512, N=1536 -> q/k/vT staging
  gemm_qkv_kernel<<<dim3(24, 392), 256, 0, stream>>>(r0, wq, qkv_b, q_s, k_s, vT_s);

  // 4) windowed attention, one block per (window, head)
  attn_kernel<<<32768, 128, 0, stream>>>(q_s, k_s, vT_s, bias_tab, r0);

  // 5) proj GEMM + bias + residual -> out (f32, unpartitioned)
  gemm_proj_kernel<<<dim3(8, 392), 256, 0, stream>>>(r0, wp, proj_b, x, out);

  // 6) LN2 -> bf16 (natural order)
  ln_to_bf16_kernel<<<12544, 256, 0, stream>>>(out, ln2_g, ln2_b, r0, 0);

  // 7) MLP1 GEMM + GELU: N=2048
  gemm_mlp1_kernel<<<dim3(32, 392), 256, 0, stream>>>(r0, w1, mlp_b1, gbuf);

  // 8) MLP2 GEMM + residual accumulate: K=2048, N=512
  gemm_mlp2_kernel<<<dim3(8, 392), 256, 0, stream>>>(gbuf, w2, mlp_b2, out);
}
